// PreTrainModel_57896159150149
// MI455X (gfx1250) — matrix operand order repaired and checked by reading, not compiled, not run on hardware
//
#include <hip/hip_runtime.h>
#include <hip/hip_bf16.h>

// Problem constants (match reference)
#define NUME   20000
#define NUMR   8
#define RREL   16          // 2*NUMR relation subgraphs
#define DIM    256
#define DEEPTH 2
#define BATCH  2048
#define NEDGE  500000
#define KPRED  (2*DIM)     // 512

typedef __bf16 bf16_t;
typedef __attribute__((ext_vector_type(16))) __bf16 v16bf;
typedef __attribute__((ext_vector_type(8)))  float  v8f;

struct AFragRaw { float4 x0, x1, y0, y1; };

// ---------------------------------------------------------------- utilities

__global__ void zero_f32_kernel(float* __restrict__ p, long long n) {
    long long i = (long long)blockIdx.x * blockDim.x + threadIdx.x;
    long long stride = (long long)gridDim.x * blockDim.x;
    for (; i < n; i += stride) p[i] = 0.0f;
}

// Pack f32 row-major [K,N] matrices (nb back-to-back) into WMMA B-fragment
// order: dst[((tile*32 + lane)*16 + j)] = (bf16)src[k,n] with
//   tile = (k0/32)*(N/16) + (n0/16), lane = hi*16 + l16,
//   k = k0 + hi*8 + (j&7) + (j>>3)*16, n = n0 + l16      (CDNA5 ISA 7.12.2, 16-bit B 32x16)
__global__ void pack_b_kernel(const float* __restrict__ src,
                              bf16_t* __restrict__ dst,
                              int K, int N, long long total) {
    long long i = (long long)blockIdx.x * blockDim.x + threadIdx.x;
    long long stride = (long long)gridDim.x * blockDim.x;
    long long per = (long long)K * N;
    int ntiles = N >> 4;
    for (; i < total; i += stride) {
        long long b = i / per;
        long long o = i - b * per;
        int tile = (int)(o >> 9);        // / (32*16)
        int lane = (int)(o >> 4) & 31;
        int j    = (int)o & 15;
        int hi = lane >> 4, l16 = lane & 15;
        int k = (tile / ntiles) * 32 + hi * 8 + (j & 7) + ((j >> 3) << 4);
        int n = (tile % ntiles) * 16 + l16;
        dst[i] = (bf16_t)src[b * per + (long long)k * N + n];
    }
}

// ------------------------------------------------------------- graph norms

__global__ void degrees_kernel(const int* __restrict__ esrc,
                               const int* __restrict__ edst,
                               const int* __restrict__ erel,
                               float* __restrict__ deg_out,
                               float* __restrict__ deg_in) {
    int i = blockIdx.x * blockDim.x + threadIdx.x;
    if (i >= NEDGE) return;
    int u = esrc[i], v = edst[i], r = erel[i];
    atomicAdd(&deg_out[(size_t)r * NUME + u], 1.0f);                 // r: u->v
    atomicAdd(&deg_in [(size_t)r * NUME + v], 1.0f);
    atomicAdd(&deg_out[(size_t)(r + NUMR) * NUME + v], 1.0f);        // -r: v->u
    atomicAdd(&deg_in [(size_t)(r + NUMR) * NUME + u], 1.0f);
}

__global__ void finalize_norm_kernel(float* __restrict__ p, long long n) {
    long long i = (long long)blockIdx.x * blockDim.x + threadIdx.x;
    if (i >= n) return;
    p[i] = rsqrtf(fmaxf(p[i], 1.0f));
}

// ------------------------------------------------------------- scatter msg

__global__ __launch_bounds__(256)
void scatter_kernel(const float* __restrict__ h,
                    const int* __restrict__ esrc,
                    const int* __restrict__ edst,
                    const int* __restrict__ erel,
                    const float* __restrict__ dnorm_out,
                    float* __restrict__ agg) {
    int wave = blockIdx.x * (blockDim.x >> 5) + (threadIdx.x >> 5);
    int lane = threadIdx.x & 31;
    if (wave >= 2 * NEDGE) return;

    int u, v, r;
    if (wave < NEDGE) {
        u = esrc[wave]; v = edst[wave]; r = erel[wave];
    } else {
        int e = wave - NEDGE;
        u = edst[e]; v = esrc[e]; r = erel[e] + NUMR;
    }
    size_t seg_src = (size_t)r * NUME + u;
    size_t seg_dst = (size_t)r * NUME + v;
    float scale = dnorm_out[seg_src];

    const float4* hp = (const float4*)(h + (size_t)u * DIM + lane * 8);
    float4 a = hp[0];
    float4 b = hp[1];
    float* ap = agg + seg_dst * DIM + lane * 8;
    atomicAdd(ap + 0, a.x * scale);
    atomicAdd(ap + 1, a.y * scale);
    atomicAdd(ap + 2, a.z * scale);
    atomicAdd(ap + 3, a.w * scale);
    atomicAdd(ap + 4, b.x * scale);
    atomicAdd(ap + 5, b.y * scale);
    atomicAdd(ap + 6, b.z * scale);
    atomicAdd(ap + 7, b.w * scale);
}

// ---------------------------------------------------- hetero-conv GEMM (WMMA)
// h_out[n,e] = (1/R) * sum_r ((agg[r,n,:]*dnorm_in[r,n]) @ W_r)[e] + meanbias[e]
// One wave per 16-row strip, 4 accumulators (16x64 output).
// 2x-unrolled ping-pong double buffer: no register copies, no loads in the
// consume phase (rowscale prefetched one iteration ahead too).

__global__ __launch_bounds__(32)
void conv_gemm_kernel(const float* __restrict__ agg,
                      const float* __restrict__ dnorm_in,
                      const bf16_t* __restrict__ Wp,   // [RREL] packed DIMxDIM
                      const float* __restrict__ bl,    // [RREL, DIM] f32
                      float* __restrict__ h_out) {
    const int lane = threadIdx.x;
    const int l16  = lane & 15;
    const int hi   = lane >> 4;
    const int m0   = blockIdx.x * 16;        // entity-row tile
    const int c0b  = blockIdx.y * 64;        // output-col strip
    const int ctile0 = c0b >> 4;
    const int NT = DIM >> 4;                 // 16 col tiles per matrix
    const int ITER = RREL * (DIM >> 5);      // 128 (even)

    v8f acc[4] = {v8f{}, v8f{}, v8f{}, v8f{}};

    auto load_a = [&](int idx, AFragRaw& f) {
        int r = idx >> 3;
        int k0 = (idx & 7) << 5;
        const float* Ab = agg + ((size_t)r * NUME + m0 + l16) * DIM + k0 + hi * 8;
        f.x0 = *(const float4*)(Ab);
        f.x1 = *(const float4*)(Ab + 4);
        f.y0 = *(const float4*)(Ab + 16);
        f.y1 = *(const float4*)(Ab + 20);
    };
    auto load_b = [&](int idx, v16bf* bF) {
        int r = idx >> 3;
        int ks = idx & 7;
        const bf16_t* Bt = Wp + (size_t)r * DIM * DIM +
                           ((size_t)(ks * NT + ctile0) * 32 + lane) * 16;
#pragma unroll
        for (int t = 0; t < 4; ++t)
            bF[t] = *(const v16bf*)(Bt + (size_t)t * 32 * 16);
    };
    auto load_rs = [&](int idx) {
        return dnorm_in[(size_t)(idx >> 3) * NUME + m0 + l16];
    };
    auto consume = [&](const AFragRaw& a, const v16bf* bF, float rs) {
        v16bf aF;
        aF[0]  = (bf16_t)(a.x0.x * rs);
        aF[1]  = (bf16_t)(a.x0.y * rs);
        aF[2]  = (bf16_t)(a.x0.z * rs);
        aF[3]  = (bf16_t)(a.x0.w * rs);
        aF[4]  = (bf16_t)(a.x1.x * rs);
        aF[5]  = (bf16_t)(a.x1.y * rs);
        aF[6]  = (bf16_t)(a.x1.z * rs);
        aF[7]  = (bf16_t)(a.x1.w * rs);
        aF[8]  = (bf16_t)(a.y0.x * rs);
        aF[9]  = (bf16_t)(a.y0.y * rs);
        aF[10] = (bf16_t)(a.y0.z * rs);
        aF[11] = (bf16_t)(a.y0.w * rs);
        aF[12] = (bf16_t)(a.y1.x * rs);
        aF[13] = (bf16_t)(a.y1.y * rs);
        aF[14] = (bf16_t)(a.y1.z * rs);
        aF[15] = (bf16_t)(a.y1.w * rs);
#pragma unroll
        for (int t = 0; t < 4; ++t)
            acc[t] = __builtin_amdgcn_wmma_f32_16x16x32_bf16(
                false, aF, false, bF[t], (short)0, acc[t], false, false);
    };

    AFragRaw aBuf0, aBuf1;
    v16bf bBuf0[4], bBuf1[4];
    float rs0, rs1;

    load_a(0, aBuf0); load_b(0, bBuf0); rs0 = load_rs(0);

    for (int idx = 0; idx < ITER; idx += 2) {
        // slot1 <- idx+1 (always valid: ITER even), then consume slot0
        load_a(idx + 1, aBuf1); load_b(idx + 1, bBuf1); rs1 = load_rs(idx + 1);
        consume(aBuf0, bBuf0, rs0);
        // slot0 <- idx+2, then consume slot1
        if (idx + 2 < ITER) {
            load_a(idx + 2, aBuf0); load_b(idx + 2, bBuf0); rs0 = load_rs(idx + 2);
        }
        consume(aBuf1, bBuf1, rs1);
    }

    const float invR = 1.0f / (float)RREL;
#pragma unroll
    for (int t = 0; t < 4; ++t) {
        int col = c0b + 16 * t + l16;
        float bm = 0.0f;
        for (int r = 0; r < RREL; ++r) bm += bl[(size_t)r * DIM + col];
        bm *= invR;
#pragma unroll
        for (int i = 0; i < 8; ++i) {
            int row = m0 + hi * 8 + i;       // C/D: VGPR i -> M = i + 8*hi
            h_out[(size_t)row * DIM + col] = acc[t][i] * invR + bm;
        }
    }
}

// ------------------------------------------------- build prediction inputs
// Writes A matrices directly in WMMA A-fragment order.

__global__ __launch_bounds__(512)
void build_pred_inputs_kernel(const float* __restrict__ h,
                              const float* __restrict__ sub_rel_emb,
                              const float* __restrict__ obj_rel_emb,
                              const int* __restrict__ sub,
                              const int* __restrict__ obj,
                              const int* __restrict__ rel,
                              bf16_t* __restrict__ A_sub,
                              bf16_t* __restrict__ A_obj) {
    int i = blockIdx.x;        // batch row
    int c = threadIdx.x;       // 0..511 (K index)
    int rl = rel[i];
    float vs, vo;
    if (c < DIM) {
        vs = h[(size_t)obj[i] * DIM + c];
        vo = h[(size_t)sub[i] * DIM + c];
    } else {
        vs = obj_rel_emb[(size_t)rl * DIM + (c - DIM)];
        vo = sub_rel_emb[(size_t)rl * DIM + (c - DIM)];
    }
    int kk = c & 31;
    int hi = (kk >> 3) & 1;
    int j  = (kk & 7) + ((kk >> 4) << 3);
    int lane = hi * 16 + (i & 15);
    size_t dst = (((size_t)(i >> 4) * (KPRED / 32) + (c >> 5)) * 32 + lane) * 16 + j;
    A_sub[dst] = (bf16_t)vs;
    A_obj[dst] = (bf16_t)vo;
}

// ------------------------------------------------------ prediction GEMM
// out[BATCH, NUME] = A[BATCH,512] @ W[512,NUME] + bias ; ping-pong pipelined.

__global__ __launch_bounds__(32)
void pred_gemm_kernel(const bf16_t* __restrict__ Ap,
                      const bf16_t* __restrict__ Wp,
                      const float* __restrict__ bias,
                      float* __restrict__ out) {
    const int lane = threadIdx.x;
    const int l16  = lane & 15;
    const int hi   = lane >> 4;
    const int m0   = blockIdx.x * 16;
    const int c0b  = blockIdx.y * 64;
    const int ctile0 = c0b >> 4;
    const int NT = NUME >> 4;                // 1250 col tiles
    const int ITER = KPRED >> 5;             // 16 k-steps (even)

    bool tvalid[4];
#pragma unroll
    for (int t = 0; t < 4; ++t) tvalid[t] = (c0b + 16 * t) < NUME;

    v8f acc[4] = {v8f{}, v8f{}, v8f{}, v8f{}};
    const bf16_t* Arow =
        Ap + ((size_t)(m0 >> 4) * (KPRED / 32) * 32 + lane) * 16;

    auto load_a = [&](int ks, v16bf& aF) {
        aF = *(const v16bf*)(Arow + (size_t)ks * 32 * 16);
    };
    auto load_b = [&](int ks, v16bf* bF) {
        const bf16_t* Bt =
            Wp + ((size_t)(ks * NT + ctile0) * 32 + lane) * 16;
#pragma unroll
        for (int t = 0; t < 4; ++t)
            if (tvalid[t]) bF[t] = *(const v16bf*)(Bt + (size_t)t * 32 * 16);
    };
    auto consume = [&](const v16bf& aF, const v16bf* bF) {
#pragma unroll
        for (int t = 0; t < 4; ++t)
            if (tvalid[t])
                acc[t] = __builtin_amdgcn_wmma_f32_16x16x32_bf16(
                    false, aF, false, bF[t], (short)0, acc[t], false, false);
    };

    v16bf aBuf0, aBuf1;
    v16bf bBuf0[4], bBuf1[4];
    load_a(0, aBuf0); load_b(0, bBuf0);

    for (int ks = 0; ks < ITER; ks += 2) {
        load_a(ks + 1, aBuf1); load_b(ks + 1, bBuf1);
        consume(aBuf0, bBuf0);
        if (ks + 2 < ITER) { load_a(ks + 2, aBuf0); load_b(ks + 2, bBuf0); }
        consume(aBuf1, bBuf1);
    }

#pragma unroll
    for (int t = 0; t < 4; ++t) {
        int col = c0b + 16 * t + l16;
        if (!tvalid[t]) continue;
        float b = bias[col];
#pragma unroll
        for (int i = 0; i < 8; ++i) {
            int row = m0 + hi * 8 + i;
            out[(size_t)row * NUME + col] = acc[t][i] + b;
        }
    }
}

// ---------------------------------------------------------------- launcher

extern "C" void kernel_launch(void* const* d_in, const int* in_sizes, int n_in,
                              void* d_out, int out_size, void* d_ws, size_t ws_size,
                              hipStream_t stream) {
    const float* entity_emb  = (const float*)d_in[0];   // [NUME, DIM]
    const float* W_conv      = (const float*)d_in[1];   // [DEEPTH, R, DIM, DIM]
    const float* b_conv      = (const float*)d_in[2];   // [DEEPTH, R, DIM]
    const float* sub_rel_emb = (const float*)d_in[3];   // [NUMR, DIM]
    const float* obj_rel_emb = (const float*)d_in[4];   // [NUMR, DIM]
    const float* W_sub       = (const float*)d_in[5];   // [512, NUME]
    const float* b_sub       = (const float*)d_in[6];   // [NUME]
    const float* W_obj       = (const float*)d_in[7];   // [512, NUME]
    const float* b_obj       = (const float*)d_in[8];   // [NUME]
    const int*   sub         = (const int*)d_in[9];     // [BATCH]
    const int*   obj         = (const int*)d_in[10];    // [BATCH]
    const int*   rel         = (const int*)d_in[11];    // [BATCH]
    const int*   edge_src    = (const int*)d_in[12];    // [NEDGE]
    const int*   edge_dst    = (const int*)d_in[13];    // [NEDGE]
    const int*   edge_rel    = (const int*)d_in[14];    // [NEDGE]
    (void)in_sizes; (void)n_in; (void)out_size; (void)ws_size;

    // ---- workspace carve-up (256B aligned slabs)
    char* base = (char*)d_ws;
    size_t off = 0;
    auto carve = [&](size_t bytes) {
        void* p = base + off;
        off += (bytes + 255) & ~(size_t)255;
        return p;
    };
    float*  dnorm_out = (float*)carve((size_t)RREL * NUME * sizeof(float));
    float*  dnorm_in  = (float*)carve((size_t)RREL * NUME * sizeof(float));
    float*  h0        = (float*)carve((size_t)NUME * DIM * sizeof(float));
    float*  h1        = (float*)carve((size_t)NUME * DIM * sizeof(float));
    float*  agg       = (float*)carve((size_t)RREL * NUME * DIM * sizeof(float));
    bf16_t* Wc16      = (bf16_t*)carve((size_t)DEEPTH * RREL * DIM * DIM * sizeof(bf16_t));
    bf16_t* Ws16      = (bf16_t*)carve((size_t)KPRED * NUME * sizeof(bf16_t));
    bf16_t* Wo16      = (bf16_t*)carve((size_t)KPRED * NUME * sizeof(bf16_t));
    bf16_t* Asub16    = (bf16_t*)carve((size_t)BATCH * KPRED * sizeof(bf16_t));
    bf16_t* Aobj16    = (bf16_t*)carve((size_t)BATCH * KPRED * sizeof(bf16_t));

    // ---- 1. degree norms (dnorm_out/dnorm_in contiguous => one zero/finalize)
    {
        long long n = 2LL * RREL * NUME;
        zero_f32_kernel<<<1024, 256, 0, stream>>>(dnorm_out, n);
        degrees_kernel<<<(NEDGE + 255) / 256, 256, 0, stream>>>(
            edge_src, edge_dst, edge_rel, dnorm_out, dnorm_in);
        finalize_norm_kernel<<<(int)((n + 255) / 256), 256, 0, stream>>>(dnorm_out, n);
    }

    // ---- 2. weights -> bf16, fragment-packed
    pack_b_kernel<<<2048, 256, 0, stream>>>(
        W_conv, Wc16, DIM, DIM, (long long)DEEPTH * RREL * DIM * DIM);
    pack_b_kernel<<<4096, 256, 0, stream>>>(
        W_sub, Ws16, KPRED, NUME, (long long)KPRED * NUME);
    pack_b_kernel<<<4096, 256, 0, stream>>>(
        W_obj, Wo16, KPRED, NUME, (long long)KPRED * NUME);

    // ---- 3. two hetero-conv layers
    const long long aggN = (long long)RREL * NUME * DIM;
    const float* hcur = entity_emb;
    float* houts[DEEPTH] = {h0, h1};
    for (int l = 0; l < DEEPTH; ++l) {
        zero_f32_kernel<<<8192, 256, 0, stream>>>(agg, aggN);
        scatter_kernel<<<(2 * NEDGE) / 8, 256, 0, stream>>>(
            hcur, edge_src, edge_dst, edge_rel, dnorm_out, agg);
        dim3 grid(NUME / 16, DIM / 64);
        conv_gemm_kernel<<<grid, 32, 0, stream>>>(
            agg, dnorm_in,
            Wc16 + (size_t)l * RREL * DIM * DIM,
            b_conv + (size_t)l * RREL * DIM,
            houts[l]);
        hcur = houts[l];
    }

    // ---- 4. prediction heads
    build_pred_inputs_kernel<<<BATCH, KPRED, 0, stream>>>(
        h1, sub_rel_emb, obj_rel_emb, sub, obj, rel, Asub16, Aobj16);

    float* out_sub = (float*)d_out;
    float* out_obj = out_sub + (size_t)BATCH * NUME;
    dim3 pgrid(BATCH / 16, (NUME / 16 + 3) / 4);
    pred_gemm_kernel<<<pgrid, 32, 0, stream>>>(Asub16, Ws16, b_sub, out_sub);
    pred_gemm_kernel<<<pgrid, 32, 0, stream>>>(Aobj16, Wo16, b_obj, out_obj);
}
